// DecoupleModel_10350871183483
// MI455X (gfx1250) — matrix operand order repaired and checked by reading, not compiled, not run on hardware
//
#include <hip/hip_runtime.h>

typedef __attribute__((ext_vector_type(16))) _Float16 v16h;
typedef __attribute__((ext_vector_type(8)))  _Float16 v8h;
typedef __attribute__((ext_vector_type(8)))  float    v8f;

#define IN_DIM       128
#define FL_DIM       512
#define OUT_DIM      64
#define ONE_PLUS_EPS 2.414213562373095f   /* 1 + sqrt(2) */
#define INV_SQRT128  0.08838834764831845f /* 1/sqrt(128) */
#define NTILE        4                    /* 16x64 output tile per wave */

#define CDIV(a, b) (((a) + (b) - 1) / (b))

// ---------------------------------------------------------------------------
// small utility kernels
// ---------------------------------------------------------------------------
__global__ void k_fill_f32(float* __restrict__ p, float v, long long n) {
    long long i = (long long)blockIdx.x * blockDim.x + threadIdx.x;
    if (i < n) p[i] = v;
}

__global__ void k_degree(const long long* __restrict__ src, float* __restrict__ norm,
                         long long E) {
    long long e = (long long)blockIdx.x * blockDim.x + threadIdx.x;
    if (e < E) atomicAdd(&norm[(int)src[e]], 1.0f);
}

__global__ void k_f32_to_f16(const float* __restrict__ x, _Float16* __restrict__ y,
                             long long n) {
    long long i = (long long)blockIdx.x * blockDim.x + threadIdx.x;
    if (i < n) y[i] = (_Float16)x[i];
}

// Transpose-convert weight W[K x Nn] (row-major f32) -> Wt[Nn x K] (row-major f16)
__global__ void k_wconv(const float* __restrict__ W, _Float16* __restrict__ Wt,
                        int K, int Nn) {
    int i = blockIdx.x * blockDim.x + threadIdx.x;
    if (i >= K * Nn) return;
    int k = i / Nn, n = i % Nn;
    Wt[(size_t)n * K + k] = (_Float16)W[i];
}

// Ah[target[e], :] += h[source[e], :]  -- one wave per edge, float4 per lane
__global__ void k_scatter(const long long* __restrict__ tgt,
                          const long long* __restrict__ src,
                          const float* __restrict__ h, float* __restrict__ Ah,
                          long long E) {
    long long wid = ((long long)blockIdx.x * blockDim.x + threadIdx.x) >> 5;
    if (wid >= E) return;
    const int lane = threadIdx.x & 31;
    const int s = (int)src[wid];
    const int t = (int)tgt[wid];
    const float4 v = *(const float4*)(h + (size_t)s * IN_DIM + lane * 4);
    float* dst = Ah + (size_t)t * IN_DIM + lane * 4;
    atomicAdd(dst + 0, v.x);
    atomicAdd(dst + 1, v.y);
    atomicAdd(dst + 2, v.z);
    atomicAdd(dst + 3, v.w);
}

// h = (1+eps)*hlin + h   (h currently holds Ah)
__global__ void k_combine(const float* __restrict__ hlin, float* __restrict__ h,
                          long long n) {
    long long i = (long long)blockIdx.x * blockDim.x + threadIdx.x;
    if (i < n) h[i] = ONE_PLUS_EPS * hlin[i] + h[i];
}

// ---------------------------------------------------------------------------
// WMMA GEMM: one wave computes one 16 x 64 output tile (4 accumulators).
//   D = epilogue( A1'@W1 [+ A2@W2] )   with A1' = RELUA ? relu(A1) : A1
// A buffers: f16 row-major [M x K]; W buffers: f16 transposed [Nn x K].
// Epilogue: t = acc + b1[n] + b2[n]; relu?; *scale; /norm[m]; -> outF / outH.
// ---------------------------------------------------------------------------
__device__ __forceinline__ v16h load_a_frag(const _Float16* __restrict__ arow,
                                            int kk, int half) {
    // A fragment (16x32): elems 0..7 -> K = half*8+e ; 8..15 -> K = 16+half*8+e
    const v8h alo = *(const v8h*)(arow + kk + half * 8);
    const v8h ahi = *(const v8h*)(arow + kk + 16 + half * 8);
    v16h a;
#pragma unroll
    for (int e = 0; e < 8; ++e) { a[e] = alo[e]; a[8 + e] = ahi[e]; }
    return a;
}

__device__ __forceinline__ v16h load_b_frag(const _Float16* __restrict__ wrow,
                                            int kk, int half) {
    // B fragment (32x16): elems 0..7 -> K = half*8+e ; 8..15 -> K = 16+half*8+e
    const v8h blo = *(const v8h*)(wrow + kk + half * 8);
    const v8h bhi = *(const v8h*)(wrow + kk + 16 + half * 8);
    v16h b;
#pragma unroll
    for (int e = 0; e < 8; ++e) { b[e] = blo[e]; b[8 + e] = bhi[e]; }
    return b;
}

template <int RELUA>
__device__ __forceinline__ void
wmma_k_loop4(const _Float16* __restrict__ arow, const _Float16* __restrict__ w0,
             const _Float16* __restrict__ w1, const _Float16* __restrict__ w2,
             const _Float16* __restrict__ w3, int K, int half, v8f acc[NTILE]) {
    for (int kk = 0; kk < K; kk += 32) {
        v16h a = load_a_frag(arow, kk, half);
        if (RELUA) {
#pragma unroll
            for (int e = 0; e < 16; ++e)
                a[e] = (a[e] > (_Float16)0) ? a[e] : (_Float16)0;
        }
        v16h b0 = load_b_frag(w0, kk, half);
        acc[0] = __builtin_amdgcn_wmma_f32_16x16x32_f16(false, a, false, b0,
                                                        (short)0, acc[0], false, false);
        v16h b1 = load_b_frag(w1, kk, half);
        acc[1] = __builtin_amdgcn_wmma_f32_16x16x32_f16(false, a, false, b1,
                                                        (short)0, acc[1], false, false);
        v16h b2 = load_b_frag(w2, kk, half);
        acc[2] = __builtin_amdgcn_wmma_f32_16x16x32_f16(false, a, false, b2,
                                                        (short)0, acc[2], false, false);
        v16h b3 = load_b_frag(w3, kk, half);
        acc[3] = __builtin_amdgcn_wmma_f32_16x16x32_f16(false, a, false, b3,
                                                        (short)0, acc[3], false, false);
    }
}

template <int RELUA, int DUAL>
__global__ void k_gemm_wmma(const _Float16* __restrict__ A1,
                            const _Float16* __restrict__ W1, int K1,
                            const _Float16* __restrict__ A2,
                            const _Float16* __restrict__ W2, int K2,
                            const float* __restrict__ b1,
                            const float* __restrict__ b2,
                            const float* __restrict__ norm, float scale,
                            int reluOut, float* __restrict__ outF,
                            _Float16* __restrict__ outH, int M, int Nn) {
    const int lane = threadIdx.x & 31;
    const int wave = threadIdx.x >> 5;
    const int ncg = Nn >> 6; // 64-wide column groups
    const int tile = blockIdx.x * (blockDim.x >> 5) + wave;
    if (tile >= (M >> 4) * ncg) return; // wave-uniform exit (EXEC stays all-1s)

    const int m0 = (tile / ncg) << 4;
    const int ng0 = (tile % ncg) << 6;
    const int half = lane >> 4;
    const int l16 = lane & 15;
    const int m = m0 + l16; // A row held by this lane

    v8f acc[NTILE] = {v8f{}, v8f{}, v8f{}, v8f{}};
    {
        const _Float16* arow = A1 + (size_t)m * K1;
        const size_t nb = (size_t)(ng0 + l16) * K1;
        const size_t st = (size_t)16 * K1;
        wmma_k_loop4<RELUA>(arow, W1 + nb, W1 + nb + st, W1 + nb + 2 * st,
                            W1 + nb + 3 * st, K1, half, acc);
    }
    if (DUAL) {
        const _Float16* arow = A2 + (size_t)m * K2;
        const size_t nb = (size_t)(ng0 + l16) * K2;
        const size_t st = (size_t)16 * K2;
        wmma_k_loop4<0>(arow, W2 + nb, W2 + nb + st, W2 + nb + 2 * st,
                        W2 + nb + 3 * st, K2, half, acc);
    }

    float nrm[8];
#pragma unroll
    for (int v = 0; v < 8; ++v)
        nrm[v] = norm ? norm[m0 + v + half * 8] : 1.0f;

#pragma unroll
    for (int j = 0; j < NTILE; ++j) {
        const int n = ng0 + j * 16 + l16; // output column held by this lane
        const float bias = (b1 ? b1[n] : 0.0f) + (b2 ? b2[n] : 0.0f);
#pragma unroll
        for (int v = 0; v < 8; ++v) {
            const int mr = m0 + v + half * 8; // C layout: VGPR v -> row v + 8*half
            float t = acc[j][v] + bias;
            if (reluOut) t = t > 0.0f ? t : 0.0f;
            t *= scale;
            t /= nrm[v];
            const size_t off = (size_t)mr * Nn + n;
            if (outF) outF[off] = t;
            if (outH) outH[off] = (_Float16)t;
        }
    }
}

// ---------------------------------------------------------------------------
// launch
// ---------------------------------------------------------------------------
extern "C" void kernel_launch(void* const* d_in, const int* in_sizes, int n_in,
                              void* d_out, int out_size, void* d_ws, size_t ws_size,
                              hipStream_t stream) {
    const float* x        = (const float*)d_in[0];
    const long long* edge = (const long long*)d_in[1]; // [2, E]: row0 target, row1 source
    const float* mpW[3] = {(const float*)d_in[2], (const float*)d_in[4], (const float*)d_in[6]};
    const float* mpb[3] = {(const float*)d_in[3], (const float*)d_in[5], (const float*)d_in[7]};
    const float* fcW0 = (const float*)d_in[8],  *fcb0 = (const float*)d_in[9];
    const float* fcW1 = (const float*)d_in[10], *fcb1 = (const float*)d_in[11];
    const float* pW0  = (const float*)d_in[12], *pb0  = (const float*)d_in[13];
    const float* pW1  = (const float*)d_in[14], *pb1  = (const float*)d_in[15];
    const float* outW = (const float*)d_in[16], *outb = (const float*)d_in[17];

    const int N       = in_sizes[0] / IN_DIM;
    const long long E = in_sizes[1] / 2;
    const long long* tgt = edge;
    const long long* src = edge + E;

    // ---- workspace partition (256B aligned) ----
    size_t off = 0;
    auto alloc = [&](size_t bytes) -> void* {
        void* r = (char*)d_ws + off;
        off = (off + bytes + 255) & ~(size_t)255;
        return r;
    };
    float*    norm = (float*)alloc((size_t)N * 4);
    float*    hA   = (float*)alloc((size_t)N * IN_DIM * 4); // h / Ah / new h
    float*    hB   = (float*)alloc((size_t)N * IN_DIM * 4); // h_lin
    _Float16* h16  = (_Float16*)alloc((size_t)N * IN_DIM * 2);
    _Float16* z16a = (_Float16*)alloc((size_t)N * FL_DIM * 2); // relu(z1)
    _Float16* z16b = (_Float16*)alloc((size_t)N * FL_DIM * 2); // z2
    _Float16* wt1  = (_Float16*)alloc((size_t)FL_DIM * FL_DIM * 2);
    _Float16* wt2  = (_Float16*)alloc((size_t)IN_DIM * FL_DIM * 2);
    (void)ws_size; (void)n_in; (void)out_size;

    const long long nh = (long long)N * IN_DIM;
    const int T = 256;
    const int WPB = T / 32; // waves (16x64 tiles) per block

    // ---- degree -> norm = 1 + eps + deg ----
    k_fill_f32<<<CDIV((long long)N, T), T, 0, stream>>>(norm, ONE_PLUS_EPS, N);
    k_degree<<<CDIV(E, (long long)T), T, 0, stream>>>(src, norm, E);

    // ---- 3 message-passing layers ----
    for (int l = 0; l < 3; ++l) {
        k_wconv<<<CDIV(IN_DIM * IN_DIM, T), T, 0, stream>>>(mpW[l], wt1, IN_DIM, IN_DIM);
        k_f32_to_f16<<<CDIV(nh, (long long)T), T, 0, stream>>>(l == 0 ? x : hA, h16, nh);
        // hB = relu(h@W + b) * (1/sqrt(128)) / norm[row]
        int tiles = (N / 16) * (IN_DIM / 64);
        k_gemm_wmma<0, 0><<<CDIV(tiles, WPB), T, 0, stream>>>(
            h16, wt1, IN_DIM, nullptr, nullptr, 0, mpb[l], nullptr, norm,
            INV_SQRT128, 1, hB, nullptr, N, IN_DIM);
        // hA = segment_sum over edges (one wave per edge)
        k_fill_f32<<<CDIV(nh, (long long)T), T, 0, stream>>>(hA, 0.0f, nh);
        k_scatter<<<CDIV(E * 32, (long long)T), T, 0, stream>>>(tgt, src, hB, hA, E);
        // hA = (1+eps)*hB + hA
        k_combine<<<CDIV(nh, (long long)T), T, 0, stream>>>(hB, hA, nh);
    }

    // x_inject = hA ; f16 copy for GEMM A-operand
    k_f32_to_f16<<<CDIV(nh, (long long)T), T, 0, stream>>>(hA, h16, nh);

    // ---- z1 = relu(h)@fcW0 + fcb0 + h@pW0 + pb0 ; store relu(z1) f16 ----
    k_wconv<<<CDIV(IN_DIM * FL_DIM, T), T, 0, stream>>>(fcW0, wt1, IN_DIM, FL_DIM);
    k_wconv<<<CDIV(IN_DIM * FL_DIM, T), T, 0, stream>>>(pW0, wt2, IN_DIM, FL_DIM);
    {
        int tiles = (N / 16) * (FL_DIM / 64);
        k_gemm_wmma<1, 1><<<CDIV(tiles, WPB), T, 0, stream>>>(
            h16, wt1, IN_DIM, h16, wt2, IN_DIM, fcb0, pb0, nullptr, 1.0f, 1,
            nullptr, z16a, N, FL_DIM);
    }

    // ---- z2 = relu(z1)@fcW1 + fcb1 + h@pW1 + pb1 ; store z2 f16 ----
    k_wconv<<<CDIV(FL_DIM * FL_DIM, T), T, 0, stream>>>(fcW1, wt1, FL_DIM, FL_DIM);
    k_wconv<<<CDIV(IN_DIM * FL_DIM, T), T, 0, stream>>>(pW1, wt2, IN_DIM, FL_DIM);
    {
        int tiles = (N / 16) * (FL_DIM / 64);
        k_gemm_wmma<0, 1><<<CDIV(tiles, WPB), T, 0, stream>>>(
            z16a, wt1, FL_DIM, h16, wt2, IN_DIM, fcb1, pb1, nullptr, 1.0f, 0,
            nullptr, z16b, N, FL_DIM);
    }

    // ---- out = z2 @ outW + outb (f32) ----
    k_wconv<<<CDIV(FL_DIM * OUT_DIM, T), T, 0, stream>>>(outW, wt1, FL_DIM, OUT_DIM);
    {
        int tiles = (N / 16) * (OUT_DIM / 64);
        k_gemm_wmma<0, 0><<<CDIV(tiles, WPB), T, 0, stream>>>(
            z16b, wt1, FL_DIM, nullptr, nullptr, 0, outb, nullptr, nullptr,
            1.0f, 0, (float*)d_out, nullptr, N, OUT_DIM);
    }
}
